// WassersteinDistance_89446988906825
// MI455X (gfx1250) — matrix operand order repaired and checked by reading, not compiled, not run on hardware
//
#include <hip/hip_runtime.h>
#include <math.h>

// Problem constants (match reference setup_inputs)
#define N_P 8192
#define N_Q 8192
#define DIM 64
#define REG_E 0.05f
#define N_ITER 20
#define LOG_A (-9.0109131f)   // -log(8192)
#define LOG_B (-9.0109131f)
#define NPART 4096

typedef __attribute__((ext_vector_type(16))) __bf16 v16bf;
typedef __attribute__((ext_vector_type(8)))  float  v8f;

// ---------------------------------------------------------------------------
// Kernel 1: row norms + init of u, v, max accumulator
// ---------------------------------------------------------------------------
__global__ __launch_bounds__(256) void k_init(const float* __restrict__ XP,
                                              const float* __restrict__ XQ,
                                              float* __restrict__ x2,
                                              float* __restrict__ y2,
                                              float* __restrict__ u,
                                              float* __restrict__ v,
                                              unsigned* __restrict__ maxbits) {
  int i = blockIdx.x * blockDim.x + threadIdx.x;
  if (i < N_P) {
    const float4* r = (const float4*)(XP + (size_t)i * DIM);
    float s = 0.f;
#pragma unroll
    for (int k = 0; k < DIM / 4; ++k) {
      float4 t = r[k];
      s += t.x * t.x + t.y * t.y + t.z * t.z + t.w * t.w;
    }
    x2[i] = s;
    u[i] = 0.f;
  }
  if (i < N_Q) {
    const float4* r = (const float4*)(XQ + (size_t)i * DIM);
    float s = 0.f;
#pragma unroll
    for (int k = 0; k < DIM / 4; ++k) {
      float4 t = r[k];
      s += t.x * t.x + t.y * t.y + t.z * t.z + t.w * t.w;
    }
    y2[i] = s;
    v[i] = 0.f;
  }
  if (i == 0) *maxbits = 0u;
}

// ---------------------------------------------------------------------------
// Kernel 2: C = max(x2 + y2 - 2*XP@XQ^T, 0) via bf16 WMMA with hi/lo split,
// fused global max reduction. Block = 8 waves, each wave one 16x16 tile,
// block covers 128 rows x 16 cols.
// ---------------------------------------------------------------------------
__global__ __launch_bounds__(256)
void k_cdist(const float* __restrict__ XP, const float* __restrict__ XQ,
             const float* __restrict__ x2, const float* __restrict__ y2,
             float* __restrict__ C, unsigned* __restrict__ maxbits) {
  const int lane = threadIdx.x & 31;
  const int wave = threadIdx.x >> 5;
  const int col0 = blockIdx.x * 16;
  const int row0 = blockIdx.y * 128 + wave * 16;

  const int half = (lane < 16) ? 0 : 1;
  const int l15  = lane & 15;

  const float* arow = XP + (size_t)(row0 + l15) * DIM;  // A: M = row0 + l15
  const float* bcol = XQ + (size_t)(col0 + l15) * DIM;  // B: N = col0 + l15

  v8f acc = {};
#pragma unroll
  for (int kc = 0; kc < DIM; kc += 32) {
    v16bf a_hi, a_lo, b_hi, b_lo;
    // A 16x32 bf16 layout (ISA 7.12.2): lanes<16 elems0-7 K=0..7, elems8-15 K=16..23
    //                                   lanes>=16 elems0-7 K=8..15, elems8-15 K=24..31
#pragma unroll
    for (int e = 0; e < 8; ++e) {
      float x0 = arow[kc + half * 8 + e];
      float x1 = arow[kc + half * 8 + e + 16];
      __bf16 h0 = (__bf16)x0;
      __bf16 h1 = (__bf16)x1;
      a_hi[e]     = h0;
      a_hi[8 + e] = h1;
      a_lo[e]     = (__bf16)(x0 - (float)h0);
      a_lo[8 + e] = (__bf16)(x1 - (float)h1);
    }
    // B 32x16 bf16 layout (same K order as A): lanes<16 elems0-7 K=0..7, elems8-15 K=16..23; lanes>=16 K=8..15, 24..31 (N = l15)
#pragma unroll
    for (int e = 0; e < 16; ++e) {
      float x = bcol[kc + half * 8 + (e & 7) + (e >> 3) * 16];
      __bf16 h = (__bf16)x;
      b_hi[e] = h;
      b_lo[e] = (__bf16)(x - (float)h);
    }
    // fp32-ish product via 3 bf16 WMMAs: hi*hi + hi*lo + lo*hi
    acc = __builtin_amdgcn_wmma_f32_16x16x32_bf16(false, a_hi, false, b_hi,
                                                  (short)0, acc, false, false);
    acc = __builtin_amdgcn_wmma_f32_16x16x32_bf16(false, a_hi, false, b_lo,
                                                  (short)0, acc, false, false);
    acc = __builtin_amdgcn_wmma_f32_16x16x32_bf16(false, a_lo, false, b_hi,
                                                  (short)0, acc, false, false);
  }

  // Epilogue: C/D layout — lane l, reg r: M = r + 8*(l>=16), N = l%16
  const int col = col0 + l15;
  const float yc = y2[col];
  float lmax = 0.f;
#pragma unroll
  for (int r = 0; r < 8; ++r) {
    int row = row0 + r + half * 8;
    float d2 = x2[row] + yc - 2.0f * acc[r];
    d2 = fmaxf(d2, 0.f);
    C[(size_t)row * N_Q + col] = d2;
    lmax = fmaxf(lmax, d2);
  }
  // max reduce (C >= 0 so float bits compare as uint)
  unsigned mb = __float_as_uint(lmax);
#pragma unroll
  for (int off = 16; off >= 1; off >>= 1) {
    unsigned o = (unsigned)__shfl_xor((int)mb, off, 32);
    mb = (o > mb) ? o : mb;
  }
  __shared__ unsigned smax[8];
  if (lane == 0) smax[wave] = mb;
  __syncthreads();
  if (threadIdx.x == 0) {
    unsigned bm = smax[0];
#pragma unroll
    for (int w = 1; w < 8; ++w) bm = (smax[w] > bm) ? smax[w] : bm;
    atomicMax(maxbits, bm);
  }
}

// ---------------------------------------------------------------------------
// Kernel 3: sinv = 1/(reg * maxC)
// ---------------------------------------------------------------------------
__global__ void k_scale(const unsigned* __restrict__ maxbits,
                        float* __restrict__ sinv) {
  if (threadIdx.x == 0 && blockIdx.x == 0)
    *sinv = 1.0f / (REG_E * __uint_as_float(*maxbits));
}

// ---------------------------------------------------------------------------
// Online-LSE helpers
// ---------------------------------------------------------------------------
__device__ __forceinline__ void lse_push(float& m, float& s, float x) {
  if (x > m) { s = 1.f + s * __expf(m - x); m = x; }
  else       { s += __expf(x - m); }
}
__device__ __forceinline__ void lse_merge(float& m, float& s, float m2, float s2) {
  if (m2 > m) { s = s2 + s * __expf(m - m2); m = m2; }
  else        { s = s + s2 * __expf(m2 - m); }
}
__device__ __forceinline__ void lse_tree(float* sm, float* ss, int tid, int off) {
  float m1 = sm[tid], s1 = ss[tid];
  float m2 = sm[tid + off], s2 = ss[tid + off];
  if (m2 > m1) {
    sm[tid] = m2;
    ss[tid] = s2 + s1 * __expf(m1 - m2);
  } else {
    ss[tid] = s1 + s2 * __expf(m2 - m1);
  }
}

// ---------------------------------------------------------------------------
// Kernel 4: u[i] = LOG_A - logsumexp_j(-C[i][j]*sinv + v[j]); one block/row.
// float4 (GLOBAL_LOAD_B128) streaming so instruction issue stays below the
// 23.3 TB/s memory roofline.
// ---------------------------------------------------------------------------
__global__ __launch_bounds__(256)
void k_row_lse(const float* __restrict__ C, const float* __restrict__ v,
               const float* __restrict__ sinv_p, float* __restrict__ u) {
  const int row = blockIdx.x;
  const float nsc = -(*sinv_p);
  const float4* crow4 = (const float4*)(C + (size_t)row * N_Q);
  const float4* v4 = (const float4*)v;
  float m = -__builtin_inff(), s = 0.f;
  for (int j = threadIdx.x; j < N_Q / 4; j += 256) {
    __builtin_prefetch(crow4 + j + 512, 0, 0);  // global_prefetch_b8 stream-ahead
    float4 cv = crow4[j];
    float4 vv = v4[j];
    lse_push(m, s, fmaf(cv.x, nsc, vv.x));
    lse_push(m, s, fmaf(cv.y, nsc, vv.y));
    lse_push(m, s, fmaf(cv.z, nsc, vv.z));
    lse_push(m, s, fmaf(cv.w, nsc, vv.w));
  }
  __shared__ float sm[256];
  __shared__ float ss[256];
  sm[threadIdx.x] = m;
  ss[threadIdx.x] = s;
  __syncthreads();
  for (int off = 128; off >= 1; off >>= 1) {
    if (threadIdx.x < off) lse_tree(sm, ss, threadIdx.x, off);
    __syncthreads();
  }
  if (threadIdx.x == 0) u[row] = LOG_A - (sm[0] + __logf(ss[0]));
}

// ---------------------------------------------------------------------------
// Kernel 5: v[j] = LOG_B - logsumexp_i(-C[i][j]*sinv + u[i]).
// Block = 64-column panel; each thread owns 4 consecutive columns (float4
// loads, coalesced 256B per 16-thread row group), 16 row-groups per panel,
// then a fixed-order LDS reduction over the row-groups (deterministic).
// ---------------------------------------------------------------------------
__global__ __launch_bounds__(256)
void k_col_lse(const float* __restrict__ C, const float* __restrict__ u,
               const float* __restrict__ sinv_p, float* __restrict__ v) {
  const int cq = threadIdx.x & 15;   // column quad within panel
  const int g  = threadIdx.x >> 4;   // row group 0..15
  const int col = blockIdx.x * 64 + cq * 4;
  const float nsc = -(*sinv_p);
  float m0 = -__builtin_inff(), m1 = m0, m2 = m0, m3 = m0;
  float s0 = 0.f, s1 = 0.f, s2 = 0.f, s3 = 0.f;
  for (int i = g; i < N_P; i += 16) {
    float4 cv = *(const float4*)(C + (size_t)i * N_Q + col);
    float ui = u[i];
    lse_push(m0, s0, fmaf(cv.x, nsc, ui));
    lse_push(m1, s1, fmaf(cv.y, nsc, ui));
    lse_push(m2, s2, fmaf(cv.z, nsc, ui));
    lse_push(m3, s3, fmaf(cv.w, nsc, ui));
  }
  __shared__ float sm[1024];  // [g][64 cols]
  __shared__ float ss[1024];
  const int b = g * 64 + cq * 4;
  sm[b + 0] = m0; ss[b + 0] = s0;
  sm[b + 1] = m1; ss[b + 1] = s1;
  sm[b + 2] = m2; ss[b + 2] = s2;
  sm[b + 3] = m3; ss[b + 3] = s3;
  __syncthreads();
  if (threadIdx.x < 64) {
    const int c = threadIdx.x;
    float m = sm[c], s = ss[c];
#pragma unroll
    for (int gg = 1; gg < 16; ++gg)
      lse_merge(m, s, sm[gg * 64 + c], ss[gg * 64 + c]);
    v[blockIdx.x * 64 + c] = LOG_B - (m + __logf(s));
  }
}

// ---------------------------------------------------------------------------
// Kernel 6: partial[b] = sum over block's contiguous chunk of C*exp(M+u+v).
// float4 streaming; a quad never crosses a row boundary (N_Q % 4 == 0).
// ---------------------------------------------------------------------------
__global__ __launch_bounds__(256)
void k_final(const float* __restrict__ C, const float* __restrict__ u,
             const float* __restrict__ v, const float* __restrict__ sinv_p,
             float* __restrict__ partial) {
  const size_t chunk4 = ((size_t)N_P * N_Q) / (4 * NPART);  // 4096 quads
  const size_t base4 = (size_t)blockIdx.x * chunk4;
  const float4* C4 = (const float4*)C;
  const float nsc = -(*sinv_p);
  float acc = 0.f;
  for (size_t t4 = base4 + threadIdx.x; t4 < base4 + chunk4; t4 += 256) {
    const size_t t = t4 * 4;
    const int i = (int)(t >> 13);      // / 8192
    const int j = (int)(t & 8191);
    const float4 cv = C4[t4];
    const float ui = u[i];
    acc += cv.x * __expf(fmaf(cv.x, nsc, ui + v[j + 0]));
    acc += cv.y * __expf(fmaf(cv.y, nsc, ui + v[j + 1]));
    acc += cv.z * __expf(fmaf(cv.z, nsc, ui + v[j + 2]));
    acc += cv.w * __expf(fmaf(cv.w, nsc, ui + v[j + 3]));
  }
  __shared__ float sr[256];
  sr[threadIdx.x] = acc;
  __syncthreads();
  for (int off = 128; off >= 1; off >>= 1) {
    if (threadIdx.x < off) sr[threadIdx.x] += sr[threadIdx.x + off];
    __syncthreads();
  }
  if (threadIdx.x == 0) partial[blockIdx.x] = sr[0];
}

// ---------------------------------------------------------------------------
// Kernel 7: deterministic fixed-order reduction of partials -> scalar out
// ---------------------------------------------------------------------------
__global__ __launch_bounds__(256)
void k_reduce(const float* __restrict__ partial, float* __restrict__ out) {
  float a = 0.f;
  for (int i = threadIdx.x; i < NPART; i += 256) a += partial[i];
  __shared__ float sr[256];
  sr[threadIdx.x] = a;
  __syncthreads();
  for (int off = 128; off >= 1; off >>= 1) {
    if (threadIdx.x < off) sr[threadIdx.x] += sr[threadIdx.x + off];
    __syncthreads();
  }
  if (threadIdx.x == 0) out[0] = sr[0];
}

// ---------------------------------------------------------------------------
extern "C" void kernel_launch(void* const* d_in, const int* in_sizes, int n_in,
                              void* d_out, int out_size, void* d_ws, size_t ws_size,
                              hipStream_t stream) {
  const float* XP = (const float*)d_in[0];
  const float* XQ = (const float*)d_in[1];
  float* out = (float*)d_out;

  const size_t NM = (size_t)N_P * N_Q;
  float* ws = (float*)d_ws;
  float* C        = ws;                       // 256 MB
  float* x2       = ws + NM;
  float* y2       = x2 + N_P;
  float* u        = y2 + N_Q;
  float* v        = u + N_P;
  unsigned* maxb  = (unsigned*)(v + N_Q);
  float* sinv     = (float*)(maxb + 1);
  float* partial  = sinv + 1;                 // NPART floats

  k_init<<<(N_P + 255) / 256, 256, 0, stream>>>(XP, XQ, x2, y2, u, v, maxb);

  dim3 gC(N_Q / 16, N_P / 128);
  k_cdist<<<gC, 256, 0, stream>>>(XP, XQ, x2, y2, C, maxb);

  k_scale<<<1, 1, 0, stream>>>(maxb, sinv);

  for (int it = 0; it < N_ITER; ++it) {
    k_row_lse<<<N_P, 256, 0, stream>>>(C, v, sinv, u);
    k_col_lse<<<N_Q / 64, 256, 0, stream>>>(C, u, sinv, v);
  }

  k_final<<<NPART, 256, 0, stream>>>(C, u, v, sinv, partial);
  k_reduce<<<1, 256, 0, stream>>>(partial, out);
}
